// MatchingNetwork_893353197967
// MI455X (gfx1250) — matrix operand order repaired and checked by reading, not compiled, not run on hardware
//
#include <hip/hip_runtime.h>

typedef __attribute__((ext_vector_type(16))) __bf16 v16bf;
typedef __attribute__((ext_vector_type(8)))  __bf16 v8bf;
typedef __attribute__((ext_vector_type(2)))  __bf16 v2bf;
typedef __attribute__((ext_vector_type(8)))  float  v8f;
typedef __attribute__((ext_vector_type(4)))  float  v4f;
typedef __attribute__((ext_vector_type(2)))  float  v2f;

#define EPB   64                 // episodes (B)
#define CCLS  20                 // classes
#define KSH   5                  // shots
#define QSH   15                 // queries per class
#define DDIM  2048               // feature dim (= hidden dim)
#define NSUP  (CCLS * KSH)       // 100 supports per episode
#define NQRY  (CCLS * QSH)       // 300 queries per episode
#define MSUP  (EPB * NSUP)       // 6400
#define MQRY  (EPB * NQRY)       // 19200

// ---------- bf16 helpers: native converts so isel can pick v_cvt_pk --------
static __device__ __forceinline__ unsigned pack2bf(float lo, float hi) {
  v2f f = {lo, hi};
  v2bf p = __builtin_convertvector(f, v2bf);   // packed f32->bf16 (RNE)
  unsigned u; __builtin_memcpy(&u, &p, 4); return u;
}
static __device__ __forceinline__ __bf16 f2bf(float f) { return (__bf16)f; }
static __device__ __forceinline__ float  bf2f(__bf16 b) { return (float)b; }

// ---------- CDNA5 async global->LDS copy (ASYNCcnt-tracked) -----------------
static __device__ __forceinline__ void async_copy_b128(unsigned lds_off,
                                                       const void* gptr) {
  unsigned long long ga = (unsigned long long)(uintptr_t)gptr;
  asm volatile("global_load_async_to_lds_b128 %0, %1, off"
               :: "v"(lds_off), "v"(ga) : "memory");
}

// ===========================================================================
// Kernel 1: Y_bf16[M, 2048] = X_f32[M, 2048] @ W_f32[2048, 2048] + bias
// 128x128 block tile, 8 waves (2M x 4N), wave tile 64x32 = 4x2 WMMA tiles.
// Register-prefetch pipeline: next K-tile's global loads are issued before the
// current tile's 8 v_wmma_f32_16x16x32_bf16 so HBM latency overlaps compute.
// f32->bf16 staging uses packed native converts; ds_store_b64 (A) / b32 (B).
// Epilogue: C tile staged in LDS, written back as coalesced b128 stores.
// ===========================================================================
__global__ __launch_bounds__(256) void encoder_gemm(
    const float* __restrict__ X, const float* __restrict__ W,
    const float* __restrict__ bias, unsigned short* __restrict__ Yout)
{
  __bf16* Y = reinterpret_cast<__bf16*>(Yout);
  __shared__ __bf16 As[128][40];   // [m][k], 80B row stride (16B aligned)
  __shared__ __bf16 Bt[128][40];   // [n][k] (W tile transposed)
  __shared__ __bf16 Cs[128][136];  // C tile staging, 272B stride (16B aligned)

  const int tid    = threadIdx.x;
  const int lane   = tid & 31;
  const int wave   = tid >> 5;
  const int waveM  = wave >> 2;        // 0..1
  const int waveN  = wave & 3;         // 0..3
  const int m0     = blockIdx.y * 128;
  const int n0     = blockIdx.x * 128;

  const int rowSel = lane & 15;
  const int hiHalf = lane >> 4;        // 0 | 1
  const int akoff  = hiHalf * 8;       // A frag K halves (ISA 16-bit A layout)
  const int bkoff  = hiHalf * 8;       // B frag K halves (same K order as A)

  v8f acc[4][2];
  const v8f zf = {0.f,0.f,0.f,0.f,0.f,0.f,0.f,0.f};
  #pragma unroll
  for (int mi = 0; mi < 4; ++mi)
    #pragma unroll
    for (int ni = 0; ni < 2; ++ni) acc[mi][ni] = zf;

  v4f ar[4];       // A prefetch regs: 4 x float4 (128x32 tile / 256 thr)
  v4f br[2][2];    // B prefetch regs: 2 items x (k-pair) x float4

  auto fetch = [&](int k0) {
    #pragma unroll
    for (int i = 0; i < 4; ++i) {                  // A: 1024 float4 chunks
      int lin = i * 256 + tid;
      int r   = lin >> 3;
      int c4  = (lin & 7) * 4;
      ar[i] = *reinterpret_cast<const v4f*>(&X[(size_t)(m0 + r) * DDIM + k0 + c4]);
    }
    #pragma unroll
    for (int i = 0; i < 2; ++i) {                  // B: 16 k-pairs x 32 n-quads
      int lin = i * 256 + tid;
      int kp  = lin >> 5;
      int c4  = (lin & 31) * 4;
      br[i][0] = *reinterpret_cast<const v4f*>(&W[(size_t)(k0 + 2*kp)     * DDIM + n0 + c4]);
      br[i][1] = *reinterpret_cast<const v4f*>(&W[(size_t)(k0 + 2*kp + 1) * DDIM + n0 + c4]);
    }
  };

  fetch(0);

  for (int k0 = 0; k0 < DDIM; k0 += 32) {
    // ---- stage prefetched regs into LDS as packed bf16 ----
    #pragma unroll
    for (int i = 0; i < 4; ++i) {
      int lin = i * 256 + tid;
      int r   = lin >> 3;
      int c4  = (lin & 7) * 4;
      uint2 u; u.x = pack2bf(ar[i][0], ar[i][1]); u.y = pack2bf(ar[i][2], ar[i][3]);
      *reinterpret_cast<uint2*>(&As[r][c4]) = u;   // one ds_store_b64
    }
    #pragma unroll
    for (int i = 0; i < 2; ++i) {
      int lin = i * 256 + tid;
      int kp  = lin >> 5;
      int c4  = (lin & 31) * 4;
      #pragma unroll
      for (int j = 0; j < 4; ++j)                  // transpose: pack (k,k+1) pair
        *reinterpret_cast<unsigned*>(&Bt[c4 + j][2*kp]) =
            pack2bf(br[i][0][j], br[i][1][j]);     // ds_store_b32
    }
    __syncthreads();

    if (k0 + 32 < DDIM) fetch(k0 + 32);            // overlap next tile's HBM loads

    #pragma unroll
    for (int mi = 0; mi < 4; ++mi) {
      int arow = waveM * 64 + mi * 16 + rowSel;
      v8bf alo = *reinterpret_cast<const v8bf*>(&As[arow][akoff]);
      v8bf ahi = *reinterpret_cast<const v8bf*>(&As[arow][16 + akoff]);
      v16bf a = __builtin_shufflevector(alo, ahi, 0,1,2,3,4,5,6,7,8,9,10,11,12,13,14,15);
      #pragma unroll
      for (int ni = 0; ni < 2; ++ni) {
        int bcol = waveN * 32 + ni * 16 + rowSel;
        v8bf blo = *reinterpret_cast<const v8bf*>(&Bt[bcol][bkoff]);
        v8bf bhi = *reinterpret_cast<const v8bf*>(&Bt[bcol][16 + bkoff]);
        v16bf bb = __builtin_shufflevector(blo, bhi, 0,1,2,3,4,5,6,7,8,9,10,11,12,13,14,15);
        acc[mi][ni] = __builtin_amdgcn_wmma_f32_16x16x32_bf16(
            false, a, false, bb, (short)0, acc[mi][ni], false, false);
      }
    }
    __syncthreads();
  }

  // ---- epilogue: stage C tile in LDS, then coalesced b128 global stores ----
  // (C frag layout: VGPR r -> M=r (lanes 0-15) / M=8+r (lanes 16-31); N=lane&15)
  #pragma unroll
  for (int ni = 0; ni < 2; ++ni) {
    int nl = waveN * 32 + ni * 16 + rowSel;
    float bv = bias[n0 + nl];
    #pragma unroll
    for (int mi = 0; mi < 4; ++mi) {
      #pragma unroll
      for (int r = 0; r < 8; ++r) {
        int ml = waveM * 64 + mi * 16 + hiHalf * 8 + r;
        Cs[ml][nl] = f2bf(acc[mi][ni][r] + bv);
      }
    }
  }
  __syncthreads();
  {
    int r    = tid >> 1;                 // 0..127
    int half = (tid & 1) * 64;           // 0 | 64
    #pragma unroll
    for (int j = 0; j < 8; ++j) {
      v8bf v = *reinterpret_cast<const v8bf*>(&Cs[r][half + j * 8]);
      *reinterpret_cast<v8bf*>(&Y[(size_t)(m0 + r) * DDIM + n0 + half + j * 8]) = v;
    }
  }
}

// ===========================================================================
// Kernel 2: inv_norm[row] = rsqrt(max(sum(s^2), 1e-10)) — wave per row
// ===========================================================================
__global__ __launch_bounds__(256) void row_inv_norm(
    const unsigned short* __restrict__ Sin, float* __restrict__ invn)
{
  const __bf16* S = reinterpret_cast<const __bf16*>(Sin);
  int wave = threadIdx.x >> 5, lane = threadIdx.x & 31;
  int row = blockIdx.x * 8 + wave;
  const __bf16* p = S + (size_t)row * DDIM;
  float s = 0.f;
  for (int i = lane; i < DDIM; i += 32) { float v = bf2f(p[i]); s += v * v; }
  #pragma unroll
  for (int off = 16; off > 0; off >>= 1) s += __shfl_xor(s, off, 32);
  if (lane == 0) invn[row] = rsqrtf(fmaxf(s, 1e-10f));
}

// ===========================================================================
// Kernel 3: per-episode sim[b, m, n] = (q_b[m] . s_b[n]) * inv_norm[b, n]
// bf16 inputs are copied global->LDS with CDNA5 async-to-LDS ops into a
// double-buffered LDS pipeline: issue buffer k+1, s_wait_asynccnt 4 (drain
// only the 4 older per-wave async ops, in-order completion), barrier, compute.
// Edge tiles use clamped addresses: garbage lands only in rows/cols whose
// outputs are never stored (M>=300 / N>=100), so all lanes always issue
// exactly 4 async ops per iteration and counter math stays exact.
// grid = (ceil(300/128)=3, 64 episodes)
// ===========================================================================
__global__ __launch_bounds__(256) void sim_gemm(
    const unsigned short* __restrict__ Qin, const unsigned short* __restrict__ Sin,
    const float* __restrict__ invn, float* __restrict__ sim)
{
  const __bf16* Qe = reinterpret_cast<const __bf16*>(Qin);
  const __bf16* Se = reinterpret_cast<const __bf16*>(Sin);
  __shared__ __bf16 AsD[2][128][40];
  __shared__ __bf16 BtD[2][128][40];

  const int tid    = threadIdx.x;
  const int lane   = tid & 31;
  const int wave   = tid >> 5;
  const int waveM  = wave >> 2;
  const int waveN  = wave & 3;
  const int m0     = blockIdx.x * 128;           // query-row tile within episode
  const int b      = blockIdx.y;
  const size_t qbase = (size_t)b * NQRY;
  const size_t sbase = (size_t)b * NSUP;

  const int rowSel = lane & 15;
  const int hiHalf = lane >> 4;
  const int akoff  = hiHalf * 8;
  const int bkoff  = hiHalf * 8;

  v8f acc[4][2];
  const v8f zf = {0.f,0.f,0.f,0.f,0.f,0.f,0.f,0.f};
  #pragma unroll
  for (int mi = 0; mi < 4; ++mi)
    #pragma unroll
    for (int ni = 0; ni < 2; ++ni) acc[mi][ni] = zf;

  // 4 async b128 ops per lane per buffer (A:2, B:2)
  auto issue_async = [&](int buf, int k0) {
    #pragma unroll
    for (int i = 0; i < 2; ++i) {                 // A tile: 512 16B chunks
      int lin = i * 256 + tid;
      int r   = lin >> 2;
      int c8  = (lin & 3) * 8;
      int gr  = (m0 + r < NQRY) ? (m0 + r) : (NQRY - 1);   // clamp
      async_copy_b128((unsigned)(uintptr_t)&AsD[buf][r][c8],
                      &Qe[(qbase + gr) * DDIM + k0 + c8]);
    }
    #pragma unroll
    for (int i = 0; i < 2; ++i) {                 // B tile: support rows
      int lin = i * 256 + tid;
      int n   = lin >> 2;
      int c8  = (lin & 3) * 8;
      int gn  = (n < NSUP) ? n : (NSUP - 1);               // clamp
      async_copy_b128((unsigned)(uintptr_t)&BtD[buf][n][c8],
                      &Se[(sbase + gn) * DDIM + k0 + c8]);
    }
  };

  issue_async(0, 0);

  for (int k0 = 0; k0 < DDIM; k0 += 32) {
    const int cur = (k0 >> 5) & 1;
    const bool more = (k0 + 32) < DDIM;
    if (more) {
      issue_async(cur ^ 1, k0 + 32);
      asm volatile("s_wait_asynccnt 0x4" ::: "memory");  // prior buffer done
    } else {
      asm volatile("s_wait_asynccnt 0x0" ::: "memory");
    }
    __syncthreads();                                      // all waves' data visible

    #pragma unroll
    for (int mi = 0; mi < 4; ++mi) {
      int arow = waveM * 64 + mi * 16 + rowSel;
      v8bf alo = *reinterpret_cast<const v8bf*>(&AsD[cur][arow][akoff]);
      v8bf ahi = *reinterpret_cast<const v8bf*>(&AsD[cur][arow][16 + akoff]);
      v16bf a = __builtin_shufflevector(alo, ahi, 0,1,2,3,4,5,6,7,8,9,10,11,12,13,14,15);
      #pragma unroll
      for (int ni = 0; ni < 2; ++ni) {
        int bcol = waveN * 32 + ni * 16 + rowSel;
        v8bf blo = *reinterpret_cast<const v8bf*>(&BtD[cur][bcol][bkoff]);
        v8bf bhi = *reinterpret_cast<const v8bf*>(&BtD[cur][bcol][16 + bkoff]);
        v16bf bb = __builtin_shufflevector(blo, bhi, 0,1,2,3,4,5,6,7,8,9,10,11,12,13,14,15);
        acc[mi][ni] = __builtin_amdgcn_wmma_f32_16x16x32_bf16(
            false, a, false, bb, (short)0, acc[mi][ni], false, false);
      }
    }
    __syncthreads();          // protect 'cur' buffer before it is re-filled
  }

  #pragma unroll
  for (int ni = 0; ni < 2; ++ni) {
    int n = waveN * 32 + ni * 16 + rowSel;
    float inw = (n < NSUP) ? invn[sbase + n] : 0.f;
    #pragma unroll
    for (int mi = 0; mi < 4; ++mi) {
      #pragma unroll
      for (int r = 0; r < 8; ++r) {
        int gm = m0 + waveM * 64 + mi * 16 + hiHalf * 8 + r;
        if (gm < NQRY && n < NSUP)
          sim[((size_t)b * NQRY + gm) * NSUP + n] = acc[mi][ni][r] * inw;
      }
    }
  }
}

// ===========================================================================
// Kernel 4: softmax over supports + one-hot weighted sum + argmax.
// One thread per query row; two passes over the 100-wide sim row (stays in L2).
// ===========================================================================
__global__ __launch_bounds__(256) void classify(
    const float* __restrict__ sim, const float* __restrict__ onehot,
    float* __restrict__ probs, float* __restrict__ pred)
{
  int row = blockIdx.x * 256 + threadIdx.x;
  if (row >= MQRY) return;
  int b = row / NQRY;
  const float* sr = sim + (size_t)row * NSUP;

  float mx = -3.0e38f;
  for (int s = 0; s < NSUP; ++s) mx = fmaxf(mx, sr[s]);

  float acc[CCLS];
  #pragma unroll
  for (int c = 0; c < CCLS; ++c) acc[c] = 0.f;
  float Z = 0.f;
  const float* oh = onehot + (size_t)b * NSUP * CCLS;
  for (int s = 0; s < NSUP; ++s) {
    float w = __expf(sr[s] - mx);
    Z += w;
    const float* o = oh + s * CCLS;
    #pragma unroll
    for (int c = 0; c < CCLS; ++c) acc[c] += w * o[c];
  }
  float inv = 1.f / Z;
  float best = -1.f; int bi = 0;
  #pragma unroll
  for (int c = 0; c < CCLS; ++c) {
    float p = acc[c] * inv;
    probs[(size_t)row * CCLS + c] = p;
    if (p > best) { best = p; bi = c; }          // strict > keeps first max
  }
  pred[row] = (float)bi;
}

// ===========================================================================
extern "C" void kernel_launch(void* const* d_in, const int* in_sizes, int n_in,
                              void* d_out, int out_size, void* d_ws, size_t ws_size,
                              hipStream_t stream) {
  const float* support = (const float*)d_in[0];   // [6400, 2048]
  const float* query   = (const float*)d_in[1];   // [19200, 2048]
  const float* onehot  = (const float*)d_in[2];   // [64, 100, 20]
  const float* W       = (const float*)d_in[3];   // [2048, 2048]
  const float* bias    = (const float*)d_in[4];   // [2048]
  float* out = (float*)d_out;                     // probs[19200*20] ++ pred[19200]

  char* ws = (char*)d_ws;
  size_t off = 0;
  unsigned short* Senc = (unsigned short*)(ws + off); off += (size_t)MSUP * DDIM * 2;
  unsigned short* Qenc = (unsigned short*)(ws + off); off += (size_t)MQRY * DDIM * 2;
  float* invn = (float*)(ws + off); off += (size_t)MSUP * 4;
  off = (off + 255) & ~(size_t)255;
  float* sim = (float*)(ws + off);  off += (size_t)EPB * NQRY * NSUP * 4;

  dim3 blk(256);
  encoder_gemm<<<dim3(DDIM / 128, MSUP / 128), blk, 0, stream>>>(support, W, bias, Senc);
  encoder_gemm<<<dim3(DDIM / 128, MQRY / 128), blk, 0, stream>>>(query, W, bias, Qenc);
  row_inv_norm<<<dim3(MSUP / 8), blk, 0, stream>>>(Senc, invn);
  sim_gemm<<<dim3((NQRY + 127) / 128, EPB), blk, 0, stream>>>(Qenc, Senc, invn, sim);
  classify<<<dim3((MQRY + 255) / 256), blk, 0, stream>>>(sim, onehot,
                                                         out, out + (size_t)MQRY * CCLS);
}
